// HighLowPassNet_84619445666489
// MI455X (gfx1250) — matrix operand order repaired and checked by reading, not compiled, not run on hardware
//
#include <hip/hip_runtime.h>
#include <math.h>
#include <stdint.h>

typedef __attribute__((ext_vector_type(16))) _Float16 v16h;
typedef __attribute__((ext_vector_type(8)))  float    v8f;

union V16 { v16h h; int4 q[2]; };

// ---------------- complex helpers ----------------
__device__ __forceinline__ float2 cadd(float2 a, float2 b){ return make_float2(a.x+b.x, a.y+b.y); }
__device__ __forceinline__ float2 csub(float2 a, float2 b){ return make_float2(a.x-b.x, a.y-b.y); }

// ---------------- batched line FFT (Stockham radix-2, LDS ping-pong) ----------------
// One block per line (row or column of one batch image). blockDim = N/2.
// dirSign = -1 forward (jnp.fft convention), +1 inverse.
// Inverse 1/N-per-pass scaling is folded into stage 0 so that both the line fill and
// the line drain run over the CDNA5 async global<->LDS DMA path (ASYNCcnt-tracked,
// no VGPR round trip):
//   global_load_async_to_lds_b64  vdst_lds, vaddr[2], off
//   global_store_async_from_lds_b64 vaddr[2], vsrc_lds, off
__global__ void fft_line_kernel(float2* __restrict__ g, int N, int logN, int isCol,
                                float dirSign, float scale) {
  extern __shared__ float2 sh[];
  float2* buf0 = sh;
  float2* buf1 = sh + N;
  int line = blockIdx.x % N;
  int b    = blockIdx.x / N;
  size_t base   = (size_t)b * N * N + (isCol ? (size_t)line : (size_t)line * N);
  size_t stride = isCol ? (size_t)N : (size_t)1;
  int t    = threadIdx.x;          // N/2 threads
  int half = N >> 1;

  uint64_t ga0 = (uint64_t)(uintptr_t)(g + base + (size_t)t * stride);
  uint64_t ga1 = (uint64_t)(uintptr_t)(g + base + (size_t)(t + half) * stride);

  // async DMA fill: global -> LDS (8B per lane per op)
  {
    unsigned l0 = (unsigned)(uintptr_t)(buf0 + t);
    unsigned l1 = (unsigned)(uintptr_t)(buf0 + t + half);
    asm volatile("global_load_async_to_lds_b64 %0, %1, off" :: "v"(l0), "v"(ga0) : "memory");
    asm volatile("global_load_async_to_lds_b64 %0, %1, off" :: "v"(l1), "v"(ga1) : "memory");
    asm volatile("s_wait_asynccnt 0" ::: "memory");
  }
  __syncthreads();

  float2* src = buf0; float2* dst = buf1;
  for (int s = 0; s < logN; ++s) {
    int sblk = 1 << s;
    int nsub = N >> s;
    int m    = nsub >> 1;
    int p = t >> s;
    int q = t & (sblk - 1);
    float2 a = src[q + sblk * p];
    float2 c = src[q + sblk * (p + m)];
    if (s == 0) {                       // fold ifft 1/N (per pass) into stage 0
      a.x *= scale; a.y *= scale; c.x *= scale; c.y *= scale;
    }
    float ang = dirSign * 6.28318530718f * (float)p / (float)nsub;
    float sv, cv; __sincosf(ang, &sv, &cv);
    float2 d = csub(a, c);
    dst[q + sblk * (2 * p)]     = cadd(a, c);
    dst[q + sblk * (2 * p + 1)] = make_float2(d.x * cv - d.y * sv, d.x * sv + d.y * cv);
    __syncthreads();
    float2* tmp = src; src = dst; dst = tmp;
  }

  // async DMA drain: LDS -> global
  {
    unsigned l0 = (unsigned)(uintptr_t)(src + t);
    unsigned l1 = (unsigned)(uintptr_t)(src + t + half);
    asm volatile("global_store_async_from_lds_b64 %0, %1, off" :: "v"(ga0), "v"(l0) : "memory");
    asm volatile("global_store_async_from_lds_b64 %0, %1, off" :: "v"(ga1), "v"(l1) : "memory");
    asm volatile("s_wait_asynccnt 0" ::: "memory");
  }
}

// ---------------- scatter / gather / pad / crop ----------------
__global__ void scatter_vis_kernel(const float* __restrict__ vr, const float* __restrict__ vi,
                                   const float* __restrict__ w, const int* __restrict__ idx,
                                   float2* __restrict__ grid, int M, int gsize) {
  int t = blockIdx.x * blockDim.x + threadIdx.x;
  int b = blockIdx.y;
  if (t >= M) return;
  float wv = w[t];
  float2* cell = grid + (size_t)b * gsize + idx[t];
  atomicAdd(&cell->x, vr[(size_t)b * M + t] * wv);
  atomicAdd(&cell->y, vi[(size_t)b * M + t] * wv);
}

// m0 = grid[idx]; sub0 = m0 * w
__global__ void gather_w_kernel(const float2* __restrict__ grid, const int* __restrict__ idx,
                                const float* __restrict__ w,
                                float2* __restrict__ m0, float2* __restrict__ sub0,
                                int gsize, int n) {
  int t = blockIdx.x * blockDim.x + threadIdx.x;
  int b = blockIdx.y;
  if (t >= n) return;
  __builtin_prefetch(idx + t + 2048, 0, 0);
  float2 v = grid[(size_t)b * gsize + idx[t]];
  m0[(size_t)b * n + t] = v;
  float wv = w[t];
  sub0[(size_t)b * n + t] = make_float2(v.x * wv, v.y * wv);
}

// out[b,i] = src[b*srcStride + idx[i]]
__global__ void gather_kernel(const float2* __restrict__ src, const int* __restrict__ idx,
                              float2* __restrict__ out, int srcStride, int n) {
  int t = blockIdx.x * blockDim.x + threadIdx.x;
  int b = blockIdx.y;
  if (t >= n) return;
  __builtin_prefetch(idx + t + 2048, 0, 0);
  out[(size_t)b * n + t] = src[(size_t)b * srcStride + idx[t]];
}

// grid[b*gsize + idx[i]] += src[b*srcStride + sel[i]]
__global__ void scatter_sel_kernel(const float2* __restrict__ src, const int* __restrict__ sel,
                                   const int* __restrict__ idx, float2* __restrict__ grid,
                                   int srcStride, int gsize, int n) {
  int t = blockIdx.x * blockDim.x + threadIdx.x;
  int b = blockIdx.y;
  if (t >= n) return;
  float2 v = src[(size_t)b * srcStride + sel[t]];
  float2* cell = grid + (size_t)b * gsize + idx[t];
  atomicAdd(&cell->x, v.x);
  atomicAdd(&cell->y, v.y);
}

// gA[idx[i]] += vals[b,i]; gB[idx[i]] += vals[b,i] - subs[b,i]
__global__ void scatter_pair_kernel(const float2* __restrict__ vals, const float2* __restrict__ subs,
                                    const int* __restrict__ idx,
                                    float2* __restrict__ gA, float2* __restrict__ gB,
                                    int gsize, int n) {
  int t = blockIdx.x * blockDim.x + threadIdx.x;
  int b = blockIdx.y;
  if (t >= n) return;
  float2 v = vals[(size_t)b * n + t];
  float2 s = subs[(size_t)b * n + t];
  float2* ca = gA + (size_t)b * gsize + idx[t];
  float2* cb = gB + (size_t)b * gsize + idx[t];
  atomicAdd(&ca->x, v.x);
  atomicAdd(&ca->y, v.y);
  atomicAdd(&cb->x, v.x - s.x);
  atomicAdd(&cb->y, v.y - s.y);
}

// dst[b*dstStride + sel[i]] = src[b,i]   (sel entries unique -> no atomics)
__global__ void set_at_kernel(float2* __restrict__ dst, const int* __restrict__ sel,
                              const float2* __restrict__ src, int dstStride, int n) {
  int t = blockIdx.x * blockDim.x + threadIdx.x;
  int b = blockIdx.y;
  if (t >= n) return;
  dst[(size_t)b * dstStride + sel[t]] = src[(size_t)b * n + t];
}

// out[b, r, c] = Re(grid[b, r, c]) for r,c < n  (grid is N x N)
__global__ void crop_real_kernel(const float2* __restrict__ grid, float* __restrict__ out,
                                 int N, int n) {
  int t = blockIdx.x * blockDim.x + threadIdx.x;   // n*n threads (n*n % 256 == 0)
  int b = blockIdx.y;
  int r = t / n, c = t % n;
  out[(size_t)b * n * n + t] = grid[(size_t)b * N * N + (size_t)r * N + c].x;
}

// grid[b, r, c] = (in[b, r, c], 0) for r,c < n   (grid pre-zeroed)
__global__ void pad_real_kernel(const float* __restrict__ in, float2* __restrict__ grid,
                                int N, int n) {
  int t = blockIdx.x * blockDim.x + threadIdx.x;
  int b = blockIdx.y;
  int r = t / n, c = t % n;
  grid[(size_t)b * N * N + (size_t)r * N + c] = make_float2(in[(size_t)b * n * n + t], 0.f);
}

// ---------------- fused conv block on WMMA --------------------------------------------
// conv3x3(2->F) -> relu/bn -> conv3x3(F->F) -> relu/bn -> conv1x1(F->1) -> sigmoid.
// One block = one 16x16 output tile of one batch image; 256 threads = 8 wave32.
//
// CDNA5 ISA 7.12.2 layouts (all K-contiguous per lane -> operands are 2x ds_load_b128):
//   A (16x32 f16):  M = lane&15; halves {ahk..ahk+7} and {16+ahk..23+ahk}, ahk = 8*(lane>=16)
//   B (32x16 f16):  N = lane&15; halves {ahk..ahk+7} and {16+ahk..23+ahk}, as A
//   C/D (16x16 f32): elem v -> row v + 8*(lane>=16), col lane&15
//
// LDS panels are [row][K] (weights transposed to [N][K]) so each operand is two aligned
// 16B vector loads.  Lifetime aliasing: sIn~sH2, im2col-panel~sW2t => <=61KB LDS.
template <int F> struct ConvLds {
  static constexpr int TS = 16, IN = 20, R1 = 18;
  static constexpr int K2  = 9 * F;
  static constexpr int K2P = ((K2 + 31) / 32) * 32;
  static constexpr int NT  = F / 16;
  static constexpr int MT1 = (R1 * R1 + 15) / 16;   // 21
  static constexpr int MT2 = (TS * TS) / 16;        // 16
  static constexpr int PROWS = MT1 * 16;            // 336 (padded conv1 rows)
  static constexpr int SZIN = IN * IN * 2 * 4;
  static constexpr int SZH2 = TS * TS * F * 2;
  static constexpr int REGA = (SZIN > SZH2) ? SZIN : SZH2;
  static constexpr int SZP1 = PROWS * 32 * 2;       // conv1 im2col panel (Kpad=32)
  static constexpr int SZW2 = F * K2P * 2;
  static constexpr int REGB = (SZP1 > SZW2) ? SZP1 : SZW2;
  static constexpr int SZH1 = PROWS * F * 2;
  static constexpr int SZW1 = F * 32 * 2;
  static constexpr int TOTAL = REGA + REGB + SZH1 + SZW1;
};

template <int F>
__launch_bounds__(256)
__global__ void conv_block_kernel(const float* __restrict__ x, const float* __restrict__ gr,
                                  const float* __restrict__ w1, const float* __restrict__ b1,
                                  const float* __restrict__ g1, const float* __restrict__ be1,
                                  const float* __restrict__ w2, const float* __restrict__ b2,
                                  const float* __restrict__ g2, const float* __restrict__ be2,
                                  const float* __restrict__ w3, const float* __restrict__ b3,
                                  float* __restrict__ out, int H) {
  using L = ConvLds<F>;
  __shared__ __align__(16) char smem[L::TOTAL];
  __shared__ float sB1[F], sS1[F], sO1[F], sB2[F], sS2[F], sO2[F], sW3[F];
  __shared__ float sB3;

  float*    sIn  = (float*)smem;                                    // region A
  _Float16* sH2  = (_Float16*)smem;                                 // region A (after conv1)
  _Float16* sP1  = (_Float16*)(smem + L::REGA);                     // region B
  _Float16* sW2t = (_Float16*)(smem + L::REGA);                     // region B (after conv1)
  _Float16* sH1  = (_Float16*)(smem + L::REGA + L::REGB);
  _Float16* sW1t = (_Float16*)(smem + L::REGA + L::REGB + L::SZH1);

  const int tid = threadIdx.x;
  const int b   = blockIdx.y;
  const int tpr = H / L::TS;
  const int ty0 = (blockIdx.x / tpr) * L::TS;
  const int tx0 = (blockIdx.x % tpr) * L::TS;
  const float bnc = 0.999500373f;   // 1/sqrt(1 + 0.001)

  // --- stage conv1 weights (transposed [f][Kpad=32]) + params + input tile (halo 2) ---
  for (int i = tid; i < F * 32; i += 256) {
    int f = i >> 5, k = i & 31;
    sW1t[i] = (_Float16)((k < 18) ? w1[k * F + f] : 0.f);   // w1: (ky,kx,ci,f) -> k=(ky*3+kx)*2+ci
  }
  for (int i = tid; i < F; i += 256) {
    sB1[i] = b1[i]; sS1[i] = g1[i] * bnc; sO1[i] = be1[i];
    sB2[i] = b2[i]; sS2[i] = g2[i] * bnc; sO2[i] = be2[i];
    sW3[i] = w3[i];
  }
  if (tid == 0) sB3 = b3[0];
  for (int i = tid; i < L::IN * L::IN; i += 256) {
    int iy = i / L::IN, ix = i % L::IN;
    int gy = ty0 + iy - 2, gx = tx0 + ix - 2;
    bool ok = (gy >= 0) && (gy < H) && (gx >= 0) && (gx < H);
    size_t gi = (size_t)b * H * H + (size_t)gy * H + gx;
    sIn[i * 2 + 0] = ok ? x[gi]  : 0.f;
    sIn[i * 2 + 1] = ok ? gr[gi] : 0.f;
  }
  __syncthreads();

  // --- build conv1 im2col panel sP1[336][32] f16 (packed half2 stores, zero-padded K) ---
  for (int i = tid; i < L::PROWS * 16; i += 256) {
    int m = i >> 4, j = i & 15;               // j = dword pair = (ky*3+kx), c in {0,1}
    float va = 0.f, vb = 0.f;
    if (m < L::R1 * L::R1 && j < 9) {
      int py = m / L::R1, px = m % L::R1;
      int pix = (py + j / 3) * L::IN + (px + j % 3);
      va = sIn[pix * 2 + 0];
      vb = sIn[pix * 2 + 1];
    }
    union { _Float16 h[2]; unsigned u; } pk;
    pk.h[0] = (_Float16)va; pk.h[1] = (_Float16)vb;
    ((unsigned*)sP1)[i] = pk.u;
  }
  __syncthreads();

  const int wave  = tid >> 5;
  const int lane  = tid & 31;
  const int ln    = lane & 15;
  const int ahk   = (lane >= 16) ? 8 : 0;     // A layout K offset (in halves)
  const int bhk   = (lane >= 16) ? 16 : 0;    // unused; B loads use ahk
  const int rowof = (lane >= 16) ? 8 : 0;     // C/D row offset
  const int aq    = ahk >> 3;                 // int4 index within a 32-half row

  // ---- conv1: 336 padded pixel-rows x F, single K step (2x b128 per operand) ----
  for (int task = wave; task < L::MT1 * L::NT; task += 8) {
    int mt = task / L::NT, nt = task % L::NT;
    V16 A, Bv;
    const int4* pr = (const int4*)(sP1 + (mt * 16 + ln) * 32);
    A.q[0] = pr[aq];
    A.q[1] = pr[2 + aq];
    const int4* pw = (const int4*)(sW1t + (nt * 16 + ln) * 32);
    Bv.q[0] = pw[aq];
    Bv.q[1] = pw[2 + aq];
    v8f C = {};
    C = __builtin_amdgcn_wmma_f32_16x16x32_f16(false, A.h, false, Bv.h, (short)0, C, false, false);
#pragma unroll
    for (int v = 0; v < 8; ++v) {            // branchless: sH1 padded to 336 rows
      int mr = mt * 16 + v + rowof;
      int ch = nt * 16 + ln;
      float val = fmaxf(C[v] + sB1[ch], 0.f) * sS1[ch] + sO1[ch];
      sH1[mr * F + ch] = (_Float16)val;
    }
  }
  __syncthreads();

  // --- stage conv2 weights transposed [f][K2P] (overwrites the im2col panel) ---
  for (int i = tid; i < F * L::K2P; i += 256) {
    int f = i / L::K2P, k = i % L::K2P;
    sW2t[i] = (_Float16)((k < L::K2) ? w2[k * F + f] : 0.f); // w2: (ky,kx,c,f) -> k=(ky*3+kx)*F+c
  }
  __syncthreads();

  // ---- conv2: 16x16 pixels x F, K = 9F in steps of 32 ----
  for (int task = wave; task < L::MT2 * L::NT; task += 8) {
    int mt = task / L::NT, nt = task % L::NT;
    int m  = mt * 16 + ln;
    int py = m >> 4, px = m & 15;
    const _Float16* wrow = sW2t + (nt * 16 + ln) * L::K2P;
    v8f C = {};
    for (int ks = 0; ks < L::K2P; ks += 32) {
      V16 A, Bv;
      if constexpr (F == 32) {
        // one (ky,kx) position per K step; channel halves {ahk..+7} and {16+ahk..+7}
        int kk = ks >> 5;
        const int4* pr = (const int4*)(sH1 + ((py + kk / 3) * L::R1 + (px + kk % 3)) * 32);
        A.q[0] = pr[aq];
        A.q[1] = pr[2 + aq];
      } else {
        // two (ky,kx) positions per K step; same channel halves {ahk..ahk+7} each
        int kk0 = ks >> 4;
        A.q[0] = ((const int4*)(sH1 + ((py + kk0 / 3) * L::R1 + (px + kk0 % 3)) * 16))[aq];
        int kk1 = kk0 + 1;
        if (kk1 < 9) {   // wave-uniform branch (ks uniform) -> no EXEC divergence
          A.q[1] = ((const int4*)(sH1 + ((py + kk1 / 3) * L::R1 + (px + kk1 % 3)) * 16))[aq];
        } else {
          A.q[1] = make_int4(0, 0, 0, 0);
        }
      }
      Bv.q[0] = *(const int4*)(wrow + ks + ahk);
      Bv.q[1] = *(const int4*)(wrow + ks + ahk + 16);
      C = __builtin_amdgcn_wmma_f32_16x16x32_f16(false, A.h, false, Bv.h, (short)0, C, false, false);
    }
#pragma unroll
    for (int v = 0; v < 8; ++v) {
      int mr = mt * 16 + v + rowof;
      int ch = nt * 16 + ln;
      float val = fmaxf(C[v] + sB2[ch], 0.f) * sS2[ch] + sO2[ch];
      sH2[mr * F + ch] = (_Float16)val;
    }
  }
  __syncthreads();

  // ---- conv3 1x1 (F->1) + sigmoid; 256 threads = 256 pixels ----
  {
    float acc = sB3;
    const _Float16* hp = sH2 + tid * F;
#pragma unroll
    for (int c = 0; c < F; ++c) acc += (float)hp[c] * sW3[c];
    float o = 1.f / (1.f + __expf(-acc));
    int py = tid / L::TS, px = tid % L::TS;
    out[(size_t)b * H * H + (size_t)(ty0 + py) * H + (tx0 + px)] = o;
  }
}

// ---------------- host side ----------------
static void run_fft2(float2* g, int N, int B, int logN, float dirSign, float scalePerPass,
                     hipStream_t s) {
  size_t shmem = 2 * (size_t)N * sizeof(float2);
  fft_line_kernel<<<dim3(N * B), N / 2, shmem, s>>>(g, N, logN, 0, dirSign, scalePerPass);
  fft_line_kernel<<<dim3(N * B), N / 2, shmem, s>>>(g, N, logN, 1, dirSign, scalePerPass);
}

extern "C" void kernel_launch(void* const* d_in, const int* in_sizes, int n_in,
                              void* d_out, int out_size, void* d_ws, size_t ws_size,
                              hipStream_t stream) {
  (void)n_in; (void)out_size; (void)ws_size;
  const float* vr = (const float*)d_in[0];
  const float* vi = (const float*)d_in[1];
  const float* w  = (const float*)d_in[2];
  const float* C0[10]; for (int i = 0; i < 10; ++i) C0[i] = (const float*)d_in[3 + i];
  const float* C1[10]; for (int i = 0; i < 10; ++i) C1[i] = (const float*)d_in[13 + i];
  const int* idx0 = (const int*)d_in[23];
  const int* idx1 = (const int*)d_in[24];
  const int* idx2 = (const int*)d_in[25];
  const int* sel1 = (const int*)d_in[26];
  const int* sel2 = (const int*)d_in[27];

  const int M  = in_sizes[2];
  const int B  = in_sizes[0] / M;
  const int n1 = in_sizes[24];
  const int n2 = in_sizes[25];

  // bump allocator over d_ws
  char* wp = (char*)d_ws;
  auto alloc = [&](size_t bytes) -> void* {
    void* r = (void*)wp;
    wp += (bytes + 255) & ~(size_t)255;
    return r;
  };
  float2* gA   = (float2*)alloc((size_t)B * 512 * 512 * sizeof(float2));
  float2* gB   = (float2*)alloc((size_t)B * 512 * 512 * sizeof(float2));
  float2* m0   = (float2*)alloc((size_t)B * M  * sizeof(float2));
  float2* sub0 = (float2*)alloc((size_t)B * M  * sizeof(float2));
  float2* sub1 = (float2*)alloc((size_t)B * n1 * sizeof(float2));
  float2* m1   = (float2*)alloc((size_t)B * n1 * sizeof(float2));
  float2* meas = (float2*)alloc((size_t)B * n1 * sizeof(float2));   // reused meas2/meas1
  float*  xb   = (float*)alloc((size_t)B * 256 * 256 * sizeof(float));
  float*  gb   = (float*)alloc((size_t)B * 256 * 256 * sizeof(float));
  float*  yb   = (float*)alloc((size_t)B * 128 * 128 * sizeof(float));

  const int TPB = 256;
  dim3 bM((M + TPB - 1) / TPB, B);
  dim3 bn1((n1 + TPB - 1) / TPB, B);
  dim3 bn2((n2 + TPB - 1) / TPB, B);
  auto zero = [&](float2* g, int N) {
    hipMemsetAsync(g, 0, (size_t)B * N * N * sizeof(float2), stream);
  };

  // S1: adj_op(vis*w, idx0, 512) -> x (256 real crop)
  zero(gA, 512);
  scatter_vis_kernel<<<bM, TPB, 0, stream>>>(vr, vi, w, idx0, gA, M, 512 * 512);
  run_fft2(gA, 512, B, 9, +1.f, 1.f / 512.f, stream);
  crop_real_kernel<<<dim3(256 * 256 / TPB, B), TPB, 0, stream>>>(gA, xb, 512, 256);

  // S2: m0 = dir_op(x, idx0, 512); sub0 = m0*w; sub1 = sub0[sel1]
  zero(gA, 512);
  pad_real_kernel<<<dim3(256 * 256 / TPB, B), TPB, 0, stream>>>(xb, gA, 512, 256);
  run_fft2(gA, 512, B, 9, -1.f, 1.f, stream);
  gather_w_kernel<<<bM, TPB, 0, stream>>>(gA, idx0, w, m0, sub0, 512 * 512, M);
  gather_kernel<<<bn1, TPB, 0, stream>>>(sub0, sel1, sub1, M, n1);

  // S3: x = adj_op(m0[sel1], idx1, 256) crop 128
  zero(gA, 256);
  scatter_sel_kernel<<<bn1, TPB, 0, stream>>>(m0, sel1, idx1, gA, M, 256 * 256, n1);
  run_fft2(gA, 256, B, 8, +1.f, 1.f / 256.f, stream);
  crop_real_kernel<<<dim3(128 * 128 / TPB, B), TPB, 0, stream>>>(gA, xb, 256, 128);

  // S4: m1 = dir_op(x, idx1, 256)
  zero(gA, 256);
  pad_real_kernel<<<dim3(128 * 128 / TPB, B), TPB, 0, stream>>>(xb, gA, 256, 128);
  run_fft2(gA, 256, B, 8, -1.f, 1.f, stream);
  gather_kernel<<<bn1, TPB, 0, stream>>>(gA, idx1, m1, 256 * 256, n1);

  // S5: 128-level: meas = dir_op(Re(adj_op(m1[sel2], idx2, 128)) crop64, idx2, 128)
  zero(gB, 128);
  scatter_sel_kernel<<<bn2, TPB, 0, stream>>>(m1, sel2, idx2, gB, n1, 128 * 128, n2);
  run_fft2(gB, 128, B, 7, +1.f, 1.f / 128.f, stream);
  crop_real_kernel<<<dim3(64 * 64 / TPB, B), TPB, 0, stream>>>(gB, xb, 128, 64);
  zero(gB, 128);
  pad_real_kernel<<<dim3(64 * 64 / TPB, B), TPB, 0, stream>>>(xb, gB, 128, 64);
  run_fft2(gB, 128, B, 7, -1.f, 1.f, stream);
  gather_kernel<<<bn2, TPB, 0, stream>>>(gB, idx2, meas, 128 * 128, n2);

  // S6: full1 = m1 with sel2 replaced (in place)
  set_at_kernel<<<bn2, TPB, 0, stream>>>(m1, sel2, meas, n1, n2);

  // S7: x = adj(full1); g = adj(full1 - sub1) at 256 -> crop 128
  zero(gA, 256); zero(gB, 256);
  scatter_pair_kernel<<<bn1, TPB, 0, stream>>>(m1, sub1, idx1, gA, gB, 256 * 256, n1);
  run_fft2(gA, 256, B, 8, +1.f, 1.f / 256.f, stream);
  run_fft2(gB, 256, B, 8, +1.f, 1.f / 256.f, stream);
  crop_real_kernel<<<dim3(128 * 128 / TPB, B), TPB, 0, stream>>>(gA, xb, 256, 128);
  crop_real_kernel<<<dim3(128 * 128 / TPB, B), TPB, 0, stream>>>(gB, gb, 256, 128);

  // S8: conv block c0 (F=32) on 128x128 -> yb   (WMMA implicit GEMM)
  conv_block_kernel<32><<<dim3(64, B), 256, 0, stream>>>(
      xb, gb, C0[0], C0[1], C0[2], C0[3], C0[4], C0[5], C0[6], C0[7], C0[8], C0[9], yb, 128);

  // S9: meas = dir_op(yb, idx1, 256); full0 = m0 with sel1 replaced
  zero(gA, 256);
  pad_real_kernel<<<dim3(128 * 128 / TPB, B), TPB, 0, stream>>>(yb, gA, 256, 128);
  run_fft2(gA, 256, B, 8, -1.f, 1.f, stream);
  gather_kernel<<<bn1, TPB, 0, stream>>>(gA, idx1, meas, 256 * 256, n1);
  set_at_kernel<<<bn1, TPB, 0, stream>>>(m0, sel1, meas, M, n1);

  // S10: x = adj(full0); g = adj(full0 - sub0) at 512 -> crop 256
  zero(gA, 512); zero(gB, 512);
  scatter_pair_kernel<<<bM, TPB, 0, stream>>>(m0, sub0, idx0, gA, gB, 512 * 512, M);
  run_fft2(gA, 512, B, 9, +1.f, 1.f / 512.f, stream);
  run_fft2(gB, 512, B, 9, +1.f, 1.f / 512.f, stream);
  crop_real_kernel<<<dim3(256 * 256 / TPB, B), TPB, 0, stream>>>(gA, xb, 512, 256);
  crop_real_kernel<<<dim3(256 * 256 / TPB, B), TPB, 0, stream>>>(gB, gb, 512, 256);

  // S11: conv block c1 (F=16) on 256x256 -> d_out
  conv_block_kernel<16><<<dim3(256, B), 256, 0, stream>>>(
      xb, gb, C1[0], C1[1], C1[2], C1[3], C1[4], C1[5], C1[6], C1[7], C1[8], C1[9],
      (float*)d_out, 256);
}
